// MatMulFreeGRU_1546188227051
// MI455X (gfx1250) — matrix operand order repaired and checked by reading, not compiled, not run on hardware
//
#include <hip/hip_runtime.h>
#include <hip/hip_bf16.h>
#include <math.h>

// ---------------------------------------------------------------------------
// GRU over N=H*W=196 steps, B=64, C=768, using v_wmma_f32_16x16x32_bf16.
// bf16 A/B fragments with fp32 accumulation; NT cache hints keep the 115MB
// one-shot x_zrn stream out of L2 so the Wh/Wx/Wp panels stay resident.
// ---------------------------------------------------------------------------

#define CB   64      // batch
#define CN   196     // sequence length (14*14)
#define CC   768     // channels
#define CD   2304    // 3*C
#define CBN  (CB*CN) // 12544 rows

typedef __attribute__((ext_vector_type(16))) __bf16 v16bf;
typedef __attribute__((ext_vector_type(8)))  float  v8f;

union V16U { v16bf v; unsigned int u[8]; };

// A fragment: 16x32 bf16 tile from row-major [*, lda] buffer.
// Contiguous-K layout used here (same as B): lanes 0-15 row M=lane, K=0..15; lanes 16-31 row
// M=lane-16, K=16..31. K-pairs are contiguous -> 32-bit loads.
__device__ __forceinline__ v16bf load_tileA(const __bf16* base, int lda,
                                            int rowBase, int kBase, int lane) {
  V16U t;
  int row = rowBase + (lane & 15);
  int hi  = (lane & 16) ? 8 : 0;
  const unsigned int* p =
      reinterpret_cast<const unsigned int*>(base + (size_t)row * lda + kBase);
#pragma unroll
  for (int v = 0; v < 8; ++v) {
    int j = v + hi;
    t.u[v] = p[j];
  }
  return t.v;
}

// B fragment: 32x16 bf16 tile where B[k][ncol] = W[dBase+ncol][kBase+k]
// (i.e. B = W^T slice, read straight out of row-major W).
// Contiguous-K layout used here (same as A): lane holds column (lane&15); lanes 0-15 K=0..15, lanes 16-31
// K=16..31, packed 2 per VGPR.
__device__ __forceinline__ v16bf load_tileB(const __bf16* W, int ldw,
                                            int dBase, int kBase, int lane) {
  V16U t;
  int d  = dBase + (lane & 15);
  int hi = (lane & 16) ? 8 : 0;
  const unsigned int* p =
      reinterpret_cast<const unsigned int*>(W + (size_t)d * ldw + kBase);
#pragma unroll
  for (int v = 0; v < 8; ++v) t.u[v] = p[v + hi];
  return t.v;
}

#define WMMA_BF16(a, b, c) \
  __builtin_amdgcn_wmma_f32_16x16x32_bf16(false, (a), false, (b), (short)0, (c), false, false)

// --------------------------- conversion / init ----------------------------

__global__ void k_f32_to_bf16(const float* __restrict__ in,
                              __bf16* __restrict__ out, int n) {
  int i = blockIdx.x * blockDim.x + threadIdx.x;
  if (i < n) out[i] = (__bf16)__builtin_nontemporal_load(in + i);
}

__global__ void k_init_h(float* __restrict__ hf32, __bf16* __restrict__ h0) {
  int i = blockIdx.x * blockDim.x + threadIdx.x;
  if (i < CB * CC) { hf32[i] = 0.0f; h0[i] = (__bf16)0.0f; }
}

// ------------------- input projection: x_zrn = x*Wx^T + bx -----------------
// grid: (18, 784); one wave per block; each wave: 1 row-tile x 8 col-tiles,
// so one A fragment feeds 8 WMMAs. x_zrn stores are non-temporal (one-shot
// 115MB stream; keep it out of L2).

__global__ __launch_bounds__(32)
void k_inproj(const __bf16* __restrict__ xbf, const __bf16* __restrict__ Wxbf,
              const float* __restrict__ bx, float* __restrict__ xzrn) {
  const int lane    = threadIdx.x;
  const int rowBase = blockIdx.y * 16;
  const int dBase0  = blockIdx.x * 128;

  v8f acc[8] = {};
  for (int kk = 0; kk < CC; kk += 32) {
    v16bf a = load_tileA(xbf, CC, rowBase, kk, lane);
    __builtin_prefetch(xbf + (size_t)(rowBase + (lane & 15)) * CC + kk + 32, 0, 1);
#pragma unroll
    for (int t = 0; t < 8; ++t) {
      v16bf b = load_tileB(Wxbf, CC, dBase0 + t * 16, kk, lane);
      acc[t] = WMMA_BF16(a, b, acc[t]);
    }
  }
  const int mOff = (lane & 16) ? 8 : 0;
  const int n    = lane & 15;
#pragma unroll
  for (int t = 0; t < 8; ++t) {
    int d = dBase0 + t * 16 + n;
    float bias = bx[d];
#pragma unroll
    for (int v = 0; v < 8; ++v) {
      int row = rowBase + v + mOff;
      __builtin_nontemporal_store(acc[t][v] + bias, xzrn + (size_t)row * CD + d);
    }
  }
}

// --------------------------- recurrent step --------------------------------
// grid: (48, 4); one wave per block. Tile (rt,ct) computes the z/r/n 16x16
// tiles of h @ Wh^T, sharing the A fragment, then applies the gates.
// x_zrn reads are non-temporal (each element consumed exactly once); the
// 3.5MB Wh panel stays L2-resident across all 196 steps.

__global__ __launch_bounds__(32)
void k_step(const __bf16* __restrict__ hcur, __bf16* __restrict__ hnext,
            float* __restrict__ hf32, const __bf16* __restrict__ Whbf,
            const float* __restrict__ bh, const float* __restrict__ xzrn,
            __bf16* __restrict__ hs, int step) {
  const int lane = threadIdx.x;
  const int ct = blockIdx.x;   // channel tile 0..47
  const int rt = blockIdx.y;   // batch tile   0..3

  v8f accZ = {}, accR = {}, accN = {};
  for (int kk = 0; kk < CC; kk += 32) {
    v16bf a  = load_tileA(hcur, CC, rt * 16, kk, lane);
    __builtin_prefetch(hcur + (size_t)(rt * 16 + (lane & 15)) * CC + kk + 32, 0, 1);
    v16bf bz = load_tileB(Whbf, CC,            ct * 16, kk, lane);
    v16bf br = load_tileB(Whbf, CC,  CC      + ct * 16, kk, lane);
    v16bf bn = load_tileB(Whbf, CC,  2 * CC  + ct * 16, kk, lane);
    accZ = WMMA_BF16(a, bz, accZ);
    accR = WMMA_BF16(a, br, accR);
    accN = WMMA_BF16(a, bn, accN);
  }

  const int mOff = (lane & 16) ? 8 : 0;
  const int c    = ct * 16 + (lane & 15);
  const float bhz = bh[c];
  const float bhr = bh[CC + c];
  const float bhn = bh[2 * CC + c];

#pragma unroll
  for (int v = 0; v < 8; ++v) {
    int b = rt * 16 + v + mOff;
    size_t xoff = ((size_t)b * CN + step) * CD;
    float xz = __builtin_nontemporal_load(xzrn + xoff + c);
    float xr = __builtin_nontemporal_load(xzrn + xoff + CC + c);
    float xn = __builtin_nontemporal_load(xzrn + xoff + 2 * CC + c);

    float z  = 1.0f / (1.0f + __expf(-(xz + accZ[v] + bhz)));
    float r  = 1.0f / (1.0f + __expf(-(xr + accR[v] + bhr)));
    float nc = tanhf(xn + r * (accN[v] + bhn));

    size_t hidx = (size_t)b * CC + c;
    float hp   = hf32[hidx];
    float hnew = (1.0f - z) * nc + z * hp;
    hf32[hidx]  = hnew;
    __bf16 hb = (__bf16)hnew;
    hnext[hidx] = hb;
    hs[((size_t)b * CN + step) * CC + c] = hb;
  }
}

// ------------------- output projection: y = hs * Wp^T ----------------------
// grid: (6, 784); one wave per block; 1 row-tile x 8 col-tiles. Final output
// stores are non-temporal (never re-read on device).

__global__ __launch_bounds__(32)
void k_outproj(const __bf16* __restrict__ hs, const __bf16* __restrict__ Wpbf,
               float* __restrict__ out) {
  const int lane    = threadIdx.x;
  const int rowBase = blockIdx.y * 16;
  const int dBase0  = blockIdx.x * 128;

  v8f acc[8] = {};
  for (int kk = 0; kk < CC; kk += 32) {
    v16bf a = load_tileA(hs, CC, rowBase, kk, lane);
    __builtin_prefetch(hs + (size_t)(rowBase + (lane & 15)) * CC + kk + 32, 0, 1);
#pragma unroll
    for (int t = 0; t < 8; ++t) {
      v16bf b = load_tileB(Wpbf, CC, dBase0 + t * 16, kk, lane);
      acc[t] = WMMA_BF16(a, b, acc[t]);
    }
  }
  const int mOff = (lane & 16) ? 8 : 0;
  const int n    = lane & 15;
#pragma unroll
  for (int t = 0; t < 8; ++t) {
    int d = dBase0 + t * 16 + n;
#pragma unroll
    for (int v = 0; v < 8; ++v) {
      int row = rowBase + v + mOff;
      __builtin_nontemporal_store(acc[t][v], out + (size_t)row * CC + d);
    }
  }
}

// ------------------------------- host side ---------------------------------

extern "C" void kernel_launch(void* const* d_in, const int* in_sizes, int n_in,
                              void* d_out, int out_size, void* d_ws, size_t ws_size,
                              hipStream_t stream) {
  const float* x  = (const float*)d_in[0];  // [B,H,W,C] = [BN, C]
  const float* Wx = (const float*)d_in[1];  // [3C, C]
  const float* bx = (const float*)d_in[2];  // [3C]
  const float* Wh = (const float*)d_in[3];  // [3C, C]
  const float* bh = (const float*)d_in[4];  // [3C]
  const float* Wp = (const float*)d_in[5];  // [C, C]
  float* out = (float*)d_out;

  // Workspace carve-up (256B aligned).
  unsigned char* base = (unsigned char*)d_ws;
  size_t off = 0;
  auto carve = [&](size_t bytes) -> unsigned char* {
    unsigned char* p = base + off;
    off = (off + bytes + 255) & ~(size_t)255;
    return p;
  };
  __bf16* xbf   = (__bf16*)carve((size_t)CBN * CC * 2);   // 19.3 MB
  __bf16* Wxbf  = (__bf16*)carve((size_t)CD * CC * 2);    // 3.5 MB
  __bf16* Whbf  = (__bf16*)carve((size_t)CD * CC * 2);    // 3.5 MB
  __bf16* Wpbf  = (__bf16*)carve((size_t)CC * CC * 2);    // 1.2 MB
  float*  xzrn  = (float*)carve((size_t)CBN * CD * 4);    // 115.6 MB
  __bf16* hsbf  = (__bf16*)carve((size_t)CBN * CC * 2);   // 19.3 MB
  float*  hf32  = (float*)carve((size_t)CB * CC * 4);
  __bf16* hb0   = (__bf16*)carve((size_t)CB * CC * 2);
  __bf16* hb1   = (__bf16*)carve((size_t)CB * CC * 2);
  (void)ws_size; (void)n_in; (void)in_sizes; (void)out_size;

  // 1) fp32 -> bf16 conversions + h init.
  {
    int n;
    n = CBN * CC; k_f32_to_bf16<<<(n + 255) / 256, 256, 0, stream>>>(x,  xbf,  n);
    n = CD  * CC; k_f32_to_bf16<<<(n + 255) / 256, 256, 0, stream>>>(Wx, Wxbf, n);
    n = CD  * CC; k_f32_to_bf16<<<(n + 255) / 256, 256, 0, stream>>>(Wh, Whbf, n);
    n = CC  * CC; k_f32_to_bf16<<<(n + 255) / 256, 256, 0, stream>>>(Wp, Wpbf, n);
    n = CB  * CC; k_init_h<<<(n + 255) / 256, 256, 0, stream>>>(hf32, hb0);
  }

  // 2) Input projection (WMMA).
  {
    dim3 grid(CD / 128, CBN / 16);  // (18, 784)
    k_inproj<<<grid, 32, 0, stream>>>(xbf, Wxbf, bx, xzrn);
  }

  // 3) Recurrent scan: 196 stream-ordered step kernels (graph-friendly).
  {
    dim3 grid(CC / 16, CB / 16);    // (48, 4)
    for (int n = 0; n < CN; ++n) {
      const __bf16* hcur = (n & 1) ? hb1 : hb0;
      __bf16*       hnxt = (n & 1) ? hb0 : hb1;
      k_step<<<grid, 32, 0, stream>>>(hcur, hnxt, hf32, Whbf, bh, xzrn, hsbf, n);
    }
  }

  // 4) Output projection (WMMA) -> fp32 d_out.
  {
    dim3 grid(CC / 128, CBN / 16);  // (6, 784)
    k_outproj<<<grid, 32, 0, stream>>>(hsbf, Wpbf, out);
  }
}
